// RelMultiHeadAttn_48232482734435
// MI455X (gfx1250) — matrix operand order repaired and checked by reading, not compiled, not run on hardware
//
#include <hip/hip_runtime.h>

// ---- problem constants ----
#define NHEAD   16
#define DHEAD   64
#define DMODEL  1024
#define QLEN    512
#define MLEN    512
#define BSZW    8
#define KLENC   1024
#define SCALE_F 0.125f      // 1/sqrt(64)
#define ROWS_Q  (QLEN * BSZW)      // 4096
#define ROWS_K  (KLENC * BSZW)     // 8192

typedef _Float16 h16;
typedef _Float16 v16h __attribute__((ext_vector_type(16)));
typedef _Float16 v8h  __attribute__((ext_vector_type(8)));
typedef float    v8f  __attribute__((ext_vector_type(8)));

// ---- WMMA fragment helpers (ISA 7.12.2 layouts, wave32) ----
// A 16x32 f16: lane<16 = row l16, elems {K0..7, K16..23}; lane>=16 same row, {K8..15, K24..31}
__device__ __forceinline__ v16h ld_a_frag(const h16* tile, int ldh, int l16, int hl) {
    union { v16h v; v8h h[2]; } a;
    const h16* p = tile + (size_t)l16 * ldh + 8 * hl;
    a.h[0] = *(const v8h*)p;
    a.h[1] = *(const v8h*)(p + 16);
    return a.v;
}
// B 32x16 f16 (stored as Bt[N][K] row-major): lane holds column n=l16, elems {K8h..8h+7, K16+8h..16+8h+7}
__device__ __forceinline__ v16h ld_b_frag(const h16* tile, int ldh, int l16, int hl) {
    union { v16h v; v8h h[2]; } b;
    const h16* p = tile + (size_t)l16 * ldh + 8 * hl;
    b.h[0] = *(const v8h*)p;
    b.h[1] = *(const v8h*)(p + 16);
    return b.v;
}
__device__ __forceinline__ v8f wmma16(v16h a, v16h b, v8f c) {
    return __builtin_amdgcn_wmma_f32_16x16x32_f16(false, a, false, b, (short)0, c, false, false);
}
static __device__ __forceinline__ v8f vzero8() {
    v8f z = {0.f,0.f,0.f,0.f,0.f,0.f,0.f,0.f};
    return z;
}

// LDS aperture maps via addr[31:0] (ISA 10.2): low 32 bits of a generic LDS pointer
// are the raw LDS byte offset required by async-to-LDS instructions.
__device__ __forceinline__ unsigned lds_off32(const void* p) {
    return (unsigned)(size_t)p;
}
// GVS async copy: 2 x B128 per call (inst offset applies to BOTH lds and global sides)
__device__ __forceinline__ void async_copy32B(unsigned lds, unsigned voff, const h16* base) {
    asm volatile("global_load_async_to_lds_b128 %0, %1, %2\n\t"
                 "global_load_async_to_lds_b128 %0, %1, %2 offset:16"
                 :: "v"(lds), "v"(voff), "s"(base) : "memory");
}
__device__ __forceinline__ void wait_async0() {
    asm volatile("s_wait_asynccnt 0" ::: "memory");
}

// ---- f32 -> f16 convert ----
__global__ __launch_bounds__(256) void relattn_f32_to_f16(const float* __restrict__ src,
                                                          h16* __restrict__ dst, int n) {
    int i = blockIdx.x * 256 + threadIdx.x;
    if (i < n) dst[i] = (h16)src[i];
}

// ---- tiled WMMA GEMM: C[M][N] = A[M][K] * B[N][K]^T (+bias[N]) ----
// block: 256 threads (8 waves), tile 128x128, K-step 32,
// double-buffered LDS filled by global_load_async_to_lds_b128 (ASYNCcnt-tracked)
__global__ __launch_bounds__(256) void relattn_gemm_f16nt(
    const h16* __restrict__ A, const h16* __restrict__ B,
    const float* __restrict__ bias, float* __restrict__ C,
    int M, int N, int K)
{
    __shared__ h16 As[2][128][40];   // +8 pad breaks LDS bank conflicts on frag reads
    __shared__ h16 Bs[2][128][40];
    const int tid  = threadIdx.x;
    const int wave = tid >> 5;
    const int lane = tid & 31;
    const int hl   = lane >> 4;
    const int l16  = lane & 15;
    const int m0 = blockIdx.y * 128;
    const int n0 = blockIdx.x * 128;
    const int wm0 = (wave >> 2) * 64;   // wave grid 2x4
    const int wn0 = (wave & 3) * 32;

    v8f acc[4][2];
#pragma unroll
    for (int i = 0; i < 4; ++i)
#pragma unroll
        for (int j = 0; j < 2; ++j) acc[i][j] = vzero8();

    const int lr = tid >> 1;             // staging row 0..127
    const int lc = (tid & 1) * 16;       // staging chunk (16 halves = 32B)
    const unsigned aoff0 = ((unsigned)(m0 + lr) * (unsigned)K + (unsigned)lc) * 2u;
    const unsigned boff0 = ((unsigned)(n0 + lr) * (unsigned)K + (unsigned)lc) * 2u;
    const unsigned ldsA0 = lds_off32(&As[0][lr][lc]);
    const unsigned ldsA1 = lds_off32(&As[1][lr][lc]);
    const unsigned ldsB0 = lds_off32(&Bs[0][lr][lc]);
    const unsigned ldsB1 = lds_off32(&Bs[1][lr][lc]);

    // prologue: stream tile k0=0 into buffer 0
    async_copy32B(ldsA0, aoff0, A);
    async_copy32B(ldsB0, boff0, B);
    wait_async0();
    __syncthreads();

    int cur = 0;
    for (int k0 = 0; k0 < K; k0 += 32) {
        // prefetch next K-tile into the other buffer while we compute
        if (k0 + 32 < K) {
            const unsigned kb = (unsigned)(k0 + 32) * 2u;
            async_copy32B(cur ? ldsA0 : ldsA1, aoff0 + kb, A);
            async_copy32B(cur ? ldsB0 : ldsB1, boff0 + kb, B);
        }

        v16h af[4], bf[2];
#pragma unroll
        for (int i = 0; i < 4; ++i) af[i] = ld_a_frag(&As[cur][wm0 + i*16][0], 40, l16, hl);
#pragma unroll
        for (int j = 0; j < 2; ++j) bf[j] = ld_b_frag(&Bs[cur][wn0 + j*16][0], 40, l16, hl);
#pragma unroll
        for (int i = 0; i < 4; ++i)
#pragma unroll
            for (int j = 0; j < 2; ++j)
                acc[i][j] = wmma16(af[i], bf[j], acc[i][j]);

        wait_async0();       // our prefetch landed in LDS
        __syncthreads();     // everyone done reading `cur`, next buffer visible to all
        cur ^= 1;
    }

#pragma unroll
    for (int i = 0; i < 4; ++i)
#pragma unroll
        for (int j = 0; j < 2; ++j) {
            const int gn = n0 + wn0 + j*16 + l16;
            const float bv = (bias != nullptr) ? bias[gn] : 0.f;
#pragma unroll
            for (int r = 0; r < 8; ++r) {
                const int gm = m0 + wm0 + i*16 + r + 8*hl;  // C/D: elem r -> row r+8*half
                C[(size_t)gm * N + gn] = acc[i][j][r] + bv;
            }
        }
}

// ---- fused rel-attention: per (64-query tile, batch, head) ----
// S = (q+rwb) K^T ; BD via R = (q+rrb) RK^T band read at m = j+511-i ; online softmax ; O += P V
#define LDA 72
constexpr unsigned ATTN_SMEM =
    (unsigned)((64*LDA*5 + 128*LDA) * sizeof(h16) + (64*64 + 64*128 + 3*64) * sizeof(float)); // 114432 B

__global__ __launch_bounds__(256) void relattn_attention(
    const float* __restrict__ q,    // [4096][1024] fp32 (row = i*8+b, col = n*64+d)
    const float* __restrict__ kv,   // [8192][2048] fp32 (row = j*8+b, k: n*128.., v: n*128+64..)
    const float* __restrict__ rk,   // [1024][1024] fp32 (row = m, col = n*64+d)
    const float* __restrict__ rwb,  // [16][64]
    const float* __restrict__ rrb,  // [16][64]
    h16* __restrict__ attn_vec)     // [4096][1024] f16
{
    extern __shared__ unsigned char smem_raw[];
    h16*   Aqrw  = (h16*)smem_raw;          // 64 x LDA   (q + r_w_bias)
    h16*   Aqrr  = Aqrw + 64*LDA;           // 64 x LDA   (q + r_r_bias)
    h16*   Kt    = Aqrr + 64*LDA;           // 64 x LDA   (rows j, cols d)
    h16*   RKt   = Kt   + 64*LDA;           // 128 x LDA  (rows m_local, cols d)
    h16*   Vt    = RKt  + 128*LDA;          // 64 x LDA   (rows d, cols j)  -- transposed
    h16*   Pb    = Vt   + 64*LDA;           // 64 x LDA   (probabilities, f16)
    float* Sf    = (float*)(Pb + 64*LDA);   // 64 x 64
    float* Rf    = Sf + 64*64;              // 64 x 128
    float* mrun  = Rf + 64*128;             // 64
    float* lrun  = mrun + 64;               // 64
    float* alphaS= lrun + 64;               // 64

    const int i0 = blockIdx.x * 64;
    const int b  = blockIdx.y;
    const int n  = blockIdx.z;
    const int tid  = threadIdx.x;
    const int wave = tid >> 5;
    const int lane = tid & 31;
    const int hl   = lane >> 4;
    const int l16  = lane & 15;

    // stage q tile + both biases (f16)
    {
        const int il = tid >> 2, ch = (tid & 3) * 16;
        const float* qrow = q + (size_t)((i0 + il) * BSZW + b) * DMODEL + n * DHEAD + ch;
        const float* bw = rwb + n * DHEAD + ch;
        const float* br = rrb + n * DHEAD + ch;
#pragma unroll
        for (int e = 0; e < 16; ++e) {
            float qv = qrow[e];
            Aqrw[il*LDA + ch + e] = (h16)(qv + bw[e]);
            Aqrr[il*LDA + ch + e] = (h16)(qv + br[e]);
        }
    }
    if (tid < 64) { mrun[tid] = -3.0e38f; lrun[tid] = 0.f; }

    v8f Oacc[2] = { vzero8(), vzero8() };   // wave owns 2 of 16 (4x4) 16x16 output subtiles

    const int njt = i0 / 64 + 9;            // last unmasked key tile: j0 = i0 + 512
    for (int jt = 0; jt < njt; ++jt) {
        const int j0 = jt * 64;
        const int mbase = j0 - i0 + 448;    // first rel-pos row needed; m_local = c - r + 63
        __syncthreads();                    // previous iter done with Kt/Vt/RKt/Pb

        // stage K, V(transposed), RK band (f16)
        {
            const int jl = tid >> 2, ch = (tid & 3) * 16;
            const float* krow = kv + (size_t)((j0 + jl) * BSZW + b) * (2*NHEAD*DHEAD) + n * 128 + ch;
            const float* vrow = krow + 64;
#pragma unroll
            for (int e = 0; e < 16; ++e) {
                Kt[jl*LDA + ch + e]   = (h16)krow[e];
                Vt[(ch + e)*LDA + jl] = (h16)vrow[e];
            }
            const int ml = tid >> 1, ch2 = (tid & 1) * 32;
            const int m = mbase + ml;
            if (m < KLENC) {
                const float* rrow = rk + (size_t)m * DMODEL + n * DHEAD + ch2;
#pragma unroll
                for (int e = 0; e < 32; ++e) RKt[ml*LDA + ch2 + e] = (h16)rrow[e];
            } else {
#pragma unroll
                for (int e = 0; e < 32; ++e) RKt[ml*LDA + ch2 + e] = (h16)0.f;
            }
        }
        __syncthreads();

        // S = Aqrw @ Kt^T : 64x64 -> 16 subtiles, 2 per wave
#pragma unroll
        for (int t = 0; t < 2; ++t) {
            const int s = wave * 2 + t, sm = s >> 2, sn = s & 3;
            v8f c = vzero8();
#pragma unroll
            for (int kk = 0; kk < 64; kk += 32)
                c = wmma16(ld_a_frag(&Aqrw[(sm*16)*LDA + kk], LDA, l16, hl),
                           ld_b_frag(&Kt  [(sn*16)*LDA + kk], LDA, l16, hl), c);
#pragma unroll
            for (int r = 0; r < 8; ++r)
                Sf[(sm*16 + r + 8*hl)*64 + sn*16 + l16] = c[r];
        }
        // R = Aqrr @ RKt^T : 64x128 -> 32 subtiles, 4 per wave
#pragma unroll
        for (int t = 0; t < 4; ++t) {
            const int s = wave * 4 + t, rm = s >> 3, rn = s & 7;
            v8f c = vzero8();
#pragma unroll
            for (int kk = 0; kk < 64; kk += 32)
                c = wmma16(ld_a_frag(&Aqrr[(rm*16)*LDA + kk], LDA, l16, hl),
                           ld_b_frag(&RKt [(rn*16)*LDA + kk], LDA, l16, hl), c);
#pragma unroll
            for (int r = 0; r < 8; ++r)
                Rf[(rm*16 + r + 8*hl)*128 + rn*16 + l16] = c[r];
        }
        __syncthreads();

        // fixup: S = (S + R[i][j-i+511 - mbase]) * SCALE, causal mask (j > i+512)
#pragma unroll
        for (int e = 0; e < 16; ++e) {
            const int idx = tid * 16 + e;
            const int r = idx >> 6, c = idx & 63;
            const int ig = i0 + r, jg = j0 + c;
            float sv = (Sf[r*64 + c] + Rf[r*128 + (c - r + 63)]) * SCALE_F;
            if (jg > ig + MLEN) sv = -1.0e30f;
            Sf[r*64 + c] = sv;
        }
        __syncthreads();

        // online softmax: threads 0..63 each own one query row
        if (tid < 64) {
            const float* srow = Sf + tid * 64;
            float mo = mrun[tid], mx = mo;
#pragma unroll
            for (int c = 0; c < 64; ++c) mx = fmaxf(mx, srow[c]);
            const float al = __expf(mo - mx);
            float sum = 0.f;
#pragma unroll
            for (int c = 0; c < 64; ++c) {
                const float p = __expf(srow[c] - mx);
                sum += p;
                Pb[tid*LDA + c] = (h16)p;
            }
            lrun[tid] = lrun[tid] * al + sum;
            mrun[tid] = mx;
            alphaS[tid] = al;
        }
        __syncthreads();

        // rescale O by alpha, then O += P @ Vt^T
#pragma unroll
        for (int t = 0; t < 2; ++t) {
            const int s = wave * 2 + t, sm = s >> 2, sn = s & 3;
            v8f c = Oacc[t];
#pragma unroll
            for (int r = 0; r < 8; ++r) c[r] *= alphaS[sm*16 + r + 8*hl];
#pragma unroll
            for (int kk = 0; kk < 64; kk += 32)
                c = wmma16(ld_a_frag(&Pb[(sm*16)*LDA + kk], LDA, l16, hl),
                           ld_b_frag(&Vt[(sn*16)*LDA + kk], LDA, l16, hl), c);
            Oacc[t] = c;
        }
    }
    __syncthreads();

    // write attn_vec (f16): row i*8+b, col n*64+d, normalized by l
#pragma unroll
    for (int t = 0; t < 2; ++t) {
        const int s = wave * 2 + t, sm = s >> 2, sn = s & 3;
#pragma unroll
        for (int r = 0; r < 8; ++r) {
            const int il = sm*16 + r + 8*hl;
            const int d  = sn*16 + l16;
            const float val = Oacc[t][r] / lrun[il];
            attn_vec[(size_t)((i0 + il) * BSZW + b) * DMODEL + n * DHEAD + d] = (h16)val;
        }
    }
}

// ---- epilogue: y = w + relu(attn_out + bo); LayerNorm(y) ----
__global__ __launch_bounds__(256) void relattn_bias_relu_ln(
    const float* __restrict__ w, const float* __restrict__ ao,
    const float* __restrict__ bo, const float* __restrict__ gamma,
    const float* __restrict__ beta, float* __restrict__ out)
{
    __shared__ float red[256];
    const int row = blockIdx.x;
    const int tid = threadIdx.x;
    const float* ar = ao + (size_t)row * DMODEL;
    const float* wr = w  + (size_t)row * DMODEL;
    float y[4];
    float s = 0.f;
#pragma unroll
    for (int e = 0; e < 4; ++e) {
        const int c = tid + e * 256;
        float v = ar[c] + bo[c];
        v = v > 0.f ? v : 0.f;
        v += wr[c];
        y[e] = v; s += v;
    }
    red[tid] = s; __syncthreads();
    for (int off = 128; off > 0; off >>= 1) { if (tid < off) red[tid] += red[tid + off]; __syncthreads(); }
    const float mu = red[0] * (1.f / (float)DMODEL);
    __syncthreads();
    float vs = 0.f;
#pragma unroll
    for (int e = 0; e < 4; ++e) { const float d = y[e] - mu; vs += d * d; }
    red[tid] = vs; __syncthreads();
    for (int off = 128; off > 0; off >>= 1) { if (tid < off) red[tid] += red[tid + off]; __syncthreads(); }
    const float inv = rsqrtf(red[0] * (1.f / (float)DMODEL) + 1e-5f);
    __syncthreads();
#pragma unroll
    for (int e = 0; e < 4; ++e) {
        const int c = tid + e * 256;
        out[(size_t)row * DMODEL + c] = (y[e] - mu) * inv * gamma[c] + beta[c];
    }
}

// ---- host orchestration ----
extern "C" void kernel_launch(void* const* d_in, const int* in_sizes, int n_in,
                              void* d_out, int out_size, void* d_ws, size_t ws_size,
                              hipStream_t stream) {
    (void)in_sizes; (void)n_in; (void)out_size; (void)ws_size;
    const float* w     = (const float*)d_in[0];
    const float* r     = (const float*)d_in[1];
    const float* mems  = (const float*)d_in[2];
    const float* rwb   = (const float*)d_in[3];
    const float* rrb   = (const float*)d_in[4];
    const float* Wq    = (const float*)d_in[5];
    const float* Wkv   = (const float*)d_in[6];
    const float* Wr    = (const float*)d_in[7];
    const float* br    = (const float*)d_in[8];
    const float* Wo    = (const float*)d_in[9];
    const float* bo    = (const float*)d_in[10];
    const float* gamma = (const float*)d_in[11];
    const float* beta  = (const float*)d_in[12];
    float* out = (float*)d_out;

    char* p = (char*)d_ws;
    auto alloc = [&](size_t bytes) -> void* {
        void* q = (void*)p;
        p += (bytes + 255) & ~(size_t)255;
        return q;
    };
    h16*   wB   = (h16*)  alloc((size_t)ROWS_Q * DMODEL * 2);
    h16*   catB = (h16*)  alloc((size_t)ROWS_K * DMODEL * 2);
    h16*   rB   = (h16*)  alloc((size_t)KLENC * DMODEL * 2);
    h16*   WqB  = (h16*)  alloc((size_t)DMODEL * DMODEL * 2);
    h16*   WkvB = (h16*)  alloc((size_t)2 * DMODEL * DMODEL * 2);
    h16*   WrB  = (h16*)  alloc((size_t)DMODEL * DMODEL * 2);
    h16*   WoB  = (h16*)  alloc((size_t)DMODEL * DMODEL * 2);
    float* qf   = (float*)alloc((size_t)ROWS_Q * DMODEL * 4);
    float* kvf  = (float*)alloc((size_t)ROWS_K * 2 * DMODEL * 4);
    float* rkf  = (float*)alloc((size_t)KLENC * DMODEL * 4);
    h16*   av   = (h16*)  alloc((size_t)ROWS_Q * DMODEL * 2);
    float* aof  = (float*)alloc((size_t)ROWS_Q * DMODEL * 4);

    auto cvt = [&](const float* src, h16* dst, int n) {
        relattn_f32_to_f16<<<(n + 255) / 256, 256, 0, stream>>>(src, dst, n);
    };
    cvt(w,    wB,                 ROWS_Q * DMODEL);
    cvt(mems, catB,               ROWS_Q * DMODEL);            // cat rows 0..4095 = mems
    cvt(w,    catB + (size_t)ROWS_Q * DMODEL, ROWS_Q * DMODEL);// cat rows 4096..8191 = w
    cvt(r,    rB,                 KLENC * DMODEL);
    cvt(Wq,   WqB,                DMODEL * DMODEL);
    cvt(Wkv,  WkvB,               2 * DMODEL * DMODEL);
    cvt(Wr,   WrB,                DMODEL * DMODEL);
    cvt(Wo,   WoB,                DMODEL * DMODEL);

    // projections (WMMA f16, f32 accumulate, async double-buffered LDS)
    relattn_gemm_f16nt<<<dim3(DMODEL/128, ROWS_Q/128), 256, 0, stream>>>(
        wB, WqB, nullptr, qf, ROWS_Q, DMODEL, DMODEL);
    relattn_gemm_f16nt<<<dim3(2*DMODEL/128, ROWS_K/128), 256, 0, stream>>>(
        catB, WkvB, nullptr, kvf, ROWS_K, 2*DMODEL, DMODEL);
    relattn_gemm_f16nt<<<dim3(DMODEL/128, KLENC/128), 256, 0, stream>>>(
        rB, WrB, br, rkf, KLENC, DMODEL, DMODEL);

    // fused relative attention (flash-style)
    relattn_attention<<<dim3(QLEN/64, BSZW, NHEAD), 256, ATTN_SMEM, stream>>>(
        qf, kvf, rkf, rwb, rrb, av);

    // output projection
    relattn_gemm_f16nt<<<dim3(DMODEL/128, ROWS_Q/128), 256, 0, stream>>>(
        av, WoB, nullptr, aof, ROWS_Q, DMODEL, DMODEL);

    // residual + relu + layernorm
    relattn_bias_relu_ln<<<ROWS_Q, 256, 0, stream>>>(w, aof, bo, gamma, beta, out);
}
